// PatchCore_67954972557814
// MI455X (gfx1250) — matrix operand order repaired and checked by reading, not compiled, not run on hardware
//
#include <hip/hip_runtime.h>
#include <hip/hip_bf16.h>

// ---------- types for WMMA ----------
typedef __bf16 bf16_t;
typedef bf16_t v16bf __attribute__((ext_vector_type(16)));
typedef float  v8f   __attribute__((ext_vector_type(8)));

union Frag { uint4 q[2]; v16bf v; };

// ---------- helpers ----------
__device__ __forceinline__ unsigned short f2bf(float f) {
  unsigned u = __float_as_uint(f);
  u += 0x7FFFu + ((u >> 16) & 1u);          // round-to-nearest-even
  return (unsigned short)(u >> 16);
}

__device__ __forceinline__ void bil_idx(int L, int O, int o, int& i0, int& i1, float& w) {
  float src = (o + 0.5f) * ((float)L / (float)O) - 0.5f;
  src = fminf(fmaxf(src, 0.0f), (float)(L - 1));
  float fl = floorf(src);
  i0 = (int)fl;
  i1 = min(i0 + 1, L - 1);
  w = src - fl;
}

// constants
#define NB     8
#define NROWS  6272      // 8*28*28
#define NMEM   20000
#define DDIM   1024

// ---------- K0: init d2min to +inf ----------
__global__ void k_init(int* __restrict__ d2min) {
  int i = blockIdx.x * 256 + threadIdx.x;
  if (i < NROWS) d2min[i] = 0x7F800000;  // +inf bits
}

// ---------- K1: memory f32 -> bf16, plus exact f32 row norms ----------
__global__ __launch_bounds__(256)
void k_mem(const float* __restrict__ mem, unsigned short* __restrict__ memB,
           float* __restrict__ mn2) {
  int lane = threadIdx.x & 31;
  int wave = threadIdx.x >> 5;
  int row  = blockIdx.x * 8 + wave;               // 2500*8 = 20000 exact
  const float* src = mem + (size_t)row * DDIM;
  unsigned short* dst = memB + (size_t)row * DDIM;
  float ss = 0.f;
#pragma unroll
  for (int i = 0; i < 8; ++i) {
    int off = lane * 4 + i * 128;
    float4 v = *(const float4*)(src + off);
    ss += v.x * v.x + v.y * v.y + v.z * v.z + v.w * v.w;
    uint2 o;
    o.x = (unsigned)f2bf(v.x) | ((unsigned)f2bf(v.y) << 16);
    o.y = (unsigned)f2bf(v.z) | ((unsigned)f2bf(v.w) << 16);
    *(uint2*)(dst + off) = o;
  }
#pragma unroll
  for (int s = 16; s >= 1; s >>= 1) ss += __shfl_xor(ss, s, 32);
  if (lane == 0) mn2[row] = ss;
}

// ---------- K2: feats dims [0,512): (box3x3(feat1) + center)/10 ----------
__global__ __launch_bounds__(512)
void k_feat1(const float* __restrict__ f1, unsigned short* __restrict__ featB,
             float* __restrict__ fn2) {
  __shared__ float red[512];
  int row = blockIdx.x;           // 0..6271
  int c   = threadIdx.x;          // 0..511
  int b = row / 784;
  int p = row % 784;
  int y = p / 28, x = p % 28;
  const float* plane = f1 + (size_t)(b * 512 + c) * 784;
  float s = 0.f;
#pragma unroll
  for (int dy = -1; dy <= 1; ++dy) {
    int yy = y + dy;
    if (yy < 0 || yy > 27) continue;
#pragma unroll
    for (int dx = -1; dx <= 1; ++dx) {
      int xx = x + dx;
      if (xx < 0 || xx > 27) continue;
      s += plane[yy * 28 + xx];
    }
  }
  s += plane[y * 28 + x];                 // center counted twice
  float v = s * 0.1f;
  featB[(size_t)row * DDIM + c] = f2bf(v);
  red[c] = v * v;
  __syncthreads();
  for (int st = 256; st > 0; st >>= 1) {
    if (c < st) red[c] += red[c + st];
    __syncthreads();
  }
  if (c == 0) fn2[row] = red[0];
}

// ---------- K3: T = box3x3(feat2[2cc]) + box3x3(feat2[2cc+1]) on 14x14 ----------
__global__ void k_box2(const float* __restrict__ f2, float* __restrict__ T) {
  int bc  = blockIdx.x;       // b*512 + cc, 0..4095
  int tid = threadIdx.x;      // 0..195
  if (tid >= 196) return;
  int y = tid / 14, x = tid % 14;
  const float* p0 = f2 + (size_t)(bc * 2) * 196;   // ch 2cc  (b*1024 + 2cc)
  const float* p1 = p0 + 196;                      // ch 2cc+1
  float s = 0.f;
#pragma unroll
  for (int dy = -1; dy <= 1; ++dy) {
    int yy = y + dy;
    if (yy < 0 || yy > 13) continue;
#pragma unroll
    for (int dx = -1; dx <= 1; ++dx) {
      int xx = x + dx;
      if (xx < 0 || xx > 13) continue;
      s += p0[yy * 14 + xx] + p1[yy * 14 + xx];
    }
  }
  T[(size_t)bc * 196 + tid] = s;
}

// ---------- K4: feats dims [512,1024): bilinear14->28(T)/18 ----------
__global__ __launch_bounds__(512)
void k_feat2(const float* __restrict__ T, unsigned short* __restrict__ featB,
             float* __restrict__ fn2) {
  __shared__ float red[512];
  int row = blockIdx.x;
  int cc  = threadIdx.x;
  int b = row / 784;
  int p = row % 784;
  int y = p / 28, x = p % 28;
  int y0, y1, x0, x1; float wy, wx;
  bil_idx(14, 28, y, y0, y1, wy);
  bil_idx(14, 28, x, x0, x1, wx);
  const float* tp = T + (size_t)(b * 512 + cc) * 196;
  float v00 = tp[y0 * 14 + x0], v01 = tp[y0 * 14 + x1];
  float v10 = tp[y1 * 14 + x0], v11 = tp[y1 * 14 + x1];
  float v = ((v00 * (1.f - wx) + v01 * wx) * (1.f - wy)
           + (v10 * (1.f - wx) + v11 * wx) * wy) * (1.f / 18.f);
  featB[(size_t)row * DDIM + 512 + cc] = f2bf(v);
  red[cc] = v * v;
  __syncthreads();
  for (int st = 256; st > 0; st >>= 1) {
    if (cc < st) red[cc] += red[cc + st];
    __syncthreads();
  }
  if (cc == 0) fn2[row] += red[0];
}

// ---------- K5: WMMA bf16 kNN GEMM + fused min-distance ----------
// wave32 per block; 4 M-tiles x 5 N-tiles per wave; K = 1024 in steps of 32.
// 9 fragment loads per 20 WMMAs -> ~36 flops per L2 byte.
__global__ __launch_bounds__(32)
void k_knn(const unsigned short* __restrict__ fB, const unsigned short* __restrict__ mB,
           const float* __restrict__ fn2, const float* __restrict__ mn2,
           int* __restrict__ d2min) {
  const int lane = threadIdx.x;
  const int lid  = lane & 15;
  const int half = lane >> 4;                 // 0 or 1
  const int m0 = blockIdx.y * 64;             // 98 * 64 = 6272
  const int n0 = blockIdx.x * 80;             // 250 * 80 = 20000

  // Single base pointer per matrix; tile offsets are compile-time immediates.
  const unsigned short* aptr = fB + (size_t)(m0 + lid) * DDIM;
  const unsigned short* bptr = mB + (size_t)(n0 + lid) * DDIM;
  const int akh = half * 8;    // A: VGPR0-3 <- k0+akh..+7 ; VGPR4-7 <- k0+16+akh..+7
  const int bkb = half * 8;    // B: VGPR0-3 <- k0+bkb..+7 ; VGPR4-7 <- k0+16+bkb..+7

  v8f acc[4][5];
#pragma unroll
  for (int mt = 0; mt < 4; ++mt)
#pragma unroll
    for (int t = 0; t < 5; ++t)
#pragma unroll
      for (int r = 0; r < 8; ++r) acc[mt][t][r] = 0.f;

  for (int k0 = 0; k0 < DDIM; k0 += 32) {
    Frag a[4];
#pragma unroll
    for (int mt = 0; mt < 4; ++mt) {
      const unsigned short* ap = aptr + mt * 16 * DDIM + k0 + akh;
      a[mt].q[0] = *(const uint4*)(ap);
      a[mt].q[1] = *(const uint4*)(ap + 16);
    }
#pragma unroll
    for (int t = 0; t < 5; ++t) {
      const unsigned short* bp = bptr + t * 16 * DDIM + k0 + bkb;
      Frag bf;
      bf.q[0] = *(const uint4*)(bp);
      bf.q[1] = *(const uint4*)(bp + 16);
#pragma unroll
      for (int mt = 0; mt < 4; ++mt)
        acc[mt][t] = __builtin_amdgcn_wmma_f32_16x16x32_bf16(
            false, a[mt].v, false, bf.v, (short)0, acc[mt][t], false, false);
    }
  }

  float mnv[5];
#pragma unroll
  for (int t = 0; t < 5; ++t) mnv[t] = mn2[n0 + t * 16 + lid];

#pragma unroll
  for (int mt = 0; mt < 4; ++mt) {
#pragma unroll
    for (int r = 0; r < 8; ++r) {
      // D layout: this element is M = m0 + 16*mt + r + 8*half, N = n0 + 16t + lid
      float mv = 1e30f;
#pragma unroll
      for (int t = 0; t < 5; ++t)
        mv = fminf(mv, mnv[t] - 2.0f * acc[mt][t][r]);
      // min across the 16 lanes sharing the same M (xor 1,2,4,8 stays in-half)
#pragma unroll
      for (int s = 1; s < 16; s <<= 1)
        mv = fminf(mv, __shfl_xor(mv, s, 32));
      if (lid == 0) {
        int m = m0 + 16 * mt + r + 8 * half;
        float d2 = fmaxf(fn2[m] + mv, 0.0f);
        atomicMin(d2min + m, __float_as_int(d2)); // nonneg float: int order == float order
      }
    }
  }
}

// ---------- K6a: per-image max score ----------
__global__ __launch_bounds__(256)
void k_imgscore(const int* __restrict__ d2min, float* __restrict__ out) {
  __shared__ float red[256];
  int b = blockIdx.x;
  float m = 0.f;
  for (int i = threadIdx.x; i < 784; i += 256)
    m = fmaxf(m, sqrtf(__int_as_float(d2min[b * 784 + i])));
  red[threadIdx.x] = m;
  __syncthreads();
  for (int st = 128; st > 0; st >>= 1) {
    if (threadIdx.x < st) red[threadIdx.x] = fmaxf(red[threadIdx.x], red[threadIdx.x + st]);
    __syncthreads();
  }
  if (threadIdx.x == 0) out[b] = red[0];
}

// ---------- K6b: bilinear 28->224 raw maps ----------
__global__ __launch_bounds__(256)
void k_rawmap(const int* __restrict__ d2min, float* __restrict__ out) {
  int idx = blockIdx.x * 256 + threadIdx.x;     // 8*224*224 = 401408
  if (idx >= 8 * 224 * 224) return;
  int b = idx / 50176;
  int rem = idx % 50176;
  int oy = rem / 224, ox = rem % 224;
  int y0, y1, x0, x1; float wy, wx;
  bil_idx(28, 224, oy, y0, y1, wy);
  bil_idx(28, 224, ox, x0, x1, wx);
  const int* base = d2min + b * 784;
  float v00 = sqrtf(__int_as_float(base[y0 * 28 + x0]));
  float v01 = sqrtf(__int_as_float(base[y0 * 28 + x1]));
  float v10 = sqrtf(__int_as_float(base[y1 * 28 + x0]));
  float v11 = sqrtf(__int_as_float(base[y1 * 28 + x1]));
  float v = (v00 * (1.f - wx) + v01 * wx) * (1.f - wy)
          + (v10 * (1.f - wx) + v11 * wx) * wy;
  out[8 + idx] = v;
}

extern "C" void kernel_launch(void* const* d_in, const int* in_sizes, int n_in,
                              void* d_out, int out_size, void* d_ws, size_t ws_size,
                              hipStream_t stream) {
  const float* feat1  = (const float*)d_in[0];   // [8,512,28,28]
  const float* feat2  = (const float*)d_in[1];   // [8,1024,14,14]
  const float* memory = (const float*)d_in[2];   // [20000,1024]
  float* out = (float*)d_out;                    // [8] scores + [8,224,224] maps
  char* ws = (char*)d_ws;

  // workspace layout (all 16B-aligned offsets)
  unsigned short* memB  = (unsigned short*)(ws + 0);          // 40,960,000 B
  unsigned short* featB = (unsigned short*)(ws + 40960000);   // 12,845,056 B
  float*          T     = (float*)(ws + 53805056);            //  3,211,264 B
  float*          mn2   = (float*)(ws + 57016320);            //     80,000 B
  float*          fn2   = (float*)(ws + 57096448);            //     25,088 B
  int*            d2min = (int*)(ws + 57121536);              //     25,088 B

  k_init <<<25, 256, 0, stream>>>(d2min);
  k_mem  <<<2500, 256, 0, stream>>>(memory, memB, mn2);
  k_feat1<<<NROWS, 512, 0, stream>>>(feat1, featB, fn2);
  k_box2 <<<8 * 512, 196, 0, stream>>>(feat2, T);
  k_feat2<<<NROWS, 512, 0, stream>>>(T, featB, fn2);
  k_knn  <<<dim3(250, 98), 32, 0, stream>>>(featB, memB, fn2, mn2, d2min);
  k_imgscore<<<NB, 256, 0, stream>>>(d2min, out);
  k_rawmap  <<<1568, 256, 0, stream>>>(d2min, out);
}
